// MultiHeadAttention_79336635891724
// MI455X (gfx1250) — matrix operand order repaired and checked by reading, not compiled, not run on hardware
//
#include <hip/hip_runtime.h>

// ---------------------------------------------------------------------------
// MultiHeadAttention for MI455X (gfx1250): bf16 WMMA (16x16x32, f32 accum),
// flash-style streaming softmax (base-2 domain), async/LDS staging of shared
// B operands, pre-transposed V, per-wave LDS transpose for P.
// ---------------------------------------------------------------------------

constexpr int NB  = 4;
constexpr int S   = 2048;
constexpr int H   = 8;
constexpr int D   = 64;
constexpr int E   = 512;          // H * D
constexpr int TOK = NB * S;       // 8192

typedef __attribute__((ext_vector_type(16))) __bf16 v16bf;
typedef __attribute__((ext_vector_type(8)))  __bf16 v8bf;
typedef __attribute__((ext_vector_type(8)))  float  v8f;

union V16 { v16bf v; uint4 u[2]; };

#define DEVINL __device__ __forceinline__

#if defined(__has_builtin)
#if __has_builtin(__builtin_amdgcn_exp2f)
#define EXP2(x) __builtin_amdgcn_exp2f(x)
#endif
#if __has_builtin(__builtin_amdgcn_global_load_async_to_lds_b128)
#define HAS_ASYNC_LDS 1
#endif
#endif
#ifndef EXP2
#define EXP2(x) exp2f(x)
#endif

#ifdef HAS_ASYNC_LDS
#if __has_builtin(__builtin_amdgcn_s_wait_asynccnt)
#define WAIT_ASYNC() __builtin_amdgcn_s_wait_asynccnt(0)
#else
#define WAIT_ASYNC() asm volatile("s_wait_asynccnt 0" ::: "memory")
#endif
// 16B memory -> LDS, ASYNCcnt-tracked, no VGPR round trip.
// Builtin signature (from hipcc diagnostic): param0 = AS(1) int-vec4*,
// i.e. (global src, lds dst, imm offset, cpol).
typedef int v4i_vs __attribute__((vector_size(16)));
DEVINL void async_cp16(unsigned short* lds_dst, const unsigned short* gsrc) {
  __builtin_amdgcn_global_load_async_to_lds_b128(
      (__attribute__((address_space(1))) v4i_vs*)gsrc,
      (__attribute__((address_space(3))) v4i_vs*)lds_dst, 0, 0);
}
#define STAGE16(dst, src) async_cp16((dst), (src))
#define STAGE_FENCE()     WAIT_ASYNC()
#else
#define STAGE16(dst, src) (*(uint4*)(dst) = *(const uint4*)(src))
#define STAGE_FENCE()     ((void)0)
#endif

// 1/sqrt(64) * log2(e): fold softmax scale into base-2 exponentials.
#define SCALE2   0.18033688011112042f
#define MASKED2  -1.8033688e8f            // (-1e9) * SCALE2

// f32 -> bf16 via v_cvt_pk_bf16_f32 (confirmed present on gfx1250).
DEVINL unsigned short bfb(float f) {
  return __builtin_bit_cast(unsigned short, (__bf16)f);
}
DEVINL uint4 cvt8(v8f f) {
  v8bf h = __builtin_convertvector(f, v8bf);
  return __builtin_bit_cast(uint4, h);
}
DEVINL v8f v8f_zero() {
  v8f z;
  #pragma unroll
  for (int i = 0; i < 8; ++i) z[i] = 0.f;
  return z;
}
DEVINL v8f wmma_bf16(const V16& a, const V16& b, v8f c) {
  return __builtin_amdgcn_wmma_f32_16x16x32_bf16(false, a.v, false, b.v,
                                                 (short)0, c, false, false);
}

// ---------------------------------------------------------------------------
// Kernel 0: convert Wo (f32 512x512) to bf16 once (8 elements / thread).
// ---------------------------------------------------------------------------
__global__ __launch_bounds__(256) void cvt_wo_kernel(
    const float* __restrict__ Wo, unsigned short* __restrict__ Wob) {
  int i = (blockIdx.x * 256 + threadIdx.x) * 8;
  *(uint4*)(Wob + i) = cvt8(*(const v8f*)(Wo + i));
}

// ---------------------------------------------------------------------------
// Kernel 1: fused Q/K/V projection. grid = (TOK/128, H, 3).
// Weight (64x64) converted to bf16 in LDS once per block (shared B operand).
// Q,K written [n,h][s][64] bf16; V written transposed [n,h][64][S] bf16.
// ---------------------------------------------------------------------------
__global__ __launch_bounds__(256) void qkv_proj_kernel(
    const float* __restrict__ Xq, const float* __restrict__ Xk,
    const float* __restrict__ Xv,
    const float* __restrict__ Wq, const float* __restrict__ bq,
    const float* __restrict__ Wk, const float* __restrict__ bk,
    const float* __restrict__ Wv, const float* __restrict__ bv,
    unsigned short* __restrict__ Qb, unsigned short* __restrict__ Kb,
    unsigned short* __restrict__ Vt) {
  __shared__ __align__(16) unsigned short wlds[64 * 72];  // stride 72: banks ok

  const int lane = threadIdx.x & 31, wave = threadIdx.x >> 5;
  const int ln = lane & 15, hs = lane >> 4;
  const int h = blockIdx.y;
  const int which = blockIdx.z;
  const float* X    = (which == 0) ? Xq : (which == 1) ? Xk : Xv;
  const float* W    = (which == 0) ? Wq : (which == 1) ? Wk : Wv;
  const float* bias = (which == 0) ? bq : (which == 1) ? bk : bv;

  // ---- stage + convert weight into LDS (cooperative, once) ----
  {
    const int r  = threadIdx.x >> 2;         // 0..63  (row e)
    const int cp = (threadIdx.x & 3) * 16;   // 0,16,32,48 (col d)
    const float* wp = W + r * 64 + cp;
    *(uint4*)&wlds[r * 72 + cp]     = cvt8(*(const v8f*)(wp));
    *(uint4*)&wlds[r * 72 + cp + 8] = cvt8(*(const v8f*)(wp + 8));
  }
  __syncthreads();

  const int rowBase = blockIdx.x * 128 + wave * 16;

  // A fragments: 16 rows x K=64 (two K=32 steps), f32 -> bf16 on the fly.
  V16 a[2];
  {
    const float* xp = X + (size_t)(rowBase + ln) * E + h * D;
    #pragma unroll
    for (int kk = 0; kk < 2; ++kk) {
      const int d0 = kk * 32 + hs * 16;
      a[kk].u[0] = cvt8(*(const v8f*)(xp + d0));
      a[kk].u[1] = cvt8(*(const v8f*)(xp + d0 + 8));
    }
  }

  v8f acc[4] = {v8f_zero(), v8f_zero(), v8f_zero(), v8f_zero()};
  #pragma unroll
  for (int kk = 0; kk < 2; ++kk) {
    V16 b[4];                 // batch all 4 B frags -> pipelined ds loads
    #pragma unroll
    for (int c = 0; c < 4; ++c) {
      const unsigned short* wp = &wlds[(c * 16 + ln) * 72 + kk * 32 + hs * 16];
      b[c].u[0] = *(const uint4*)(wp);
      b[c].u[1] = *(const uint4*)(wp + 8);
    }
    #pragma unroll
    for (int c = 0; c < 4; ++c)
      acc[c] = wmma_bf16(a[kk], b[c], acc[c]);
  }

  const int n = rowBase / S;
  const int sBase = rowBase % S;
  const size_t nh = (size_t)n * H + h;

  if (which < 2) {
    unsigned short* dst = (which == 0) ? Qb : Kb;
    #pragma unroll
    for (int c = 0; c < 4; ++c) {
      const int e = c * 16 + ln;
      const float bb = bias[e];
      #pragma unroll
      for (int j = 0; j < 8; ++j) {     // C layout: row = j + hs*8, col = ln
        const int s = sBase + j + hs * 8;
        dst[(nh * S + s) * D + e] = bfb(acc[c][j] + bb);
      }
    }
  } else {
    // V transposed: per lane the 8 C rows are contiguous s -> one 16B store.
    #pragma unroll
    for (int c = 0; c < 4; ++c) {
      const int e = c * 16 + ln;
      v8f t = acc[c] + bias[e];
      *(uint4*)(Vt + (nh * D + e) * S + sBase + hs * 8) = cvt8(t);
    }
  }
}

// ---------------------------------------------------------------------------
// Kernel 2: streaming (flash) attention. grid = (S/128, NB*H).
// All 8 waves of a block share (n,h): stage 64-key K/V chunks in LDS once
// per block (async direct-to-LDS when available).
// ---------------------------------------------------------------------------
__global__ __launch_bounds__(256) void attn_kernel(
    const unsigned short* __restrict__ Qb, const unsigned short* __restrict__ Kb,
    const unsigned short* __restrict__ Vt, const int* __restrict__ mask,
    unsigned short* __restrict__ Ab) {
  __shared__ __align__(16) unsigned short ks[64 * 72];   // [kpos][d], padded
  __shared__ __align__(16) unsigned short vs[64 * 72];   // [d][kpos], padded
  __shared__ __align__(16) unsigned short plds[8 * 16 * 40];  // per-wave P^T

  const int lane = threadIdx.x & 31, wave = threadIdx.x >> 5;
  const int ln = lane & 15, hs = lane >> 4;
  const int nh = blockIdx.y;
  const int n = nh / H, h = nh % H;
  const int qBase = blockIdx.x * 128 + wave * 16;
  const int wbase = wave * (16 * 40);

  // Q A-fragments (16 x 64), kept in registers for the whole key loop.
  V16 qa[2];
  {
    const unsigned short* qp = Qb + ((size_t)nh * S + qBase + ln) * D;
    #pragma unroll
    for (int kk = 0; kk < 2; ++kk) {
      qa[kk].u[0] = *(const uint4*)(qp + kk * 32 + hs * 16);
      qa[kk].u[1] = *(const uint4*)(qp + kk * 32 + hs * 16 + 8);
    }
  }

  v8f acc[4] = {v8f_zero(), v8f_zero(), v8f_zero(), v8f_zero()};
  float mrow[8], lrow[8];
  #pragma unroll
  for (int j = 0; j < 8; ++j) { mrow[j] = -3.0e38f; lrow[j] = 0.f; }

  const unsigned short* kbase = Kb + (size_t)nh * S * D;
  const unsigned short* vbase = Vt + (size_t)nh * D * S;
  const int* mk = mask + n * S;

  const int cr = threadIdx.x >> 2;          // 0..63 staging row
  const int cc = (threadIdx.x & 3) * 16;    // staging col part

  for (int kb = 0; kb < S; kb += 64) {
    __syncthreads();                         // prior chunk fully consumed
    {
      const unsigned short* ksrc = kbase + (size_t)(kb + cr) * D + cc;
      STAGE16(&ks[cr * 72 + cc],     ksrc);
      STAGE16(&ks[cr * 72 + cc + 8], ksrc + 8);
      const unsigned short* vsrc = vbase + (size_t)cr * S + kb + cc;
      STAGE16(&vs[cr * 72 + cc],     vsrc);
      STAGE16(&vs[cr * 72 + cc + 8], vsrc + 8);
      STAGE_FENCE();                         // this wave's async copies done
    }
    __syncthreads();                         // chunk visible to all waves

    #pragma unroll
    for (int kc = 0; kc < 64; kc += 32) {
      // ---- scores: 16 q-rows x 32 keys (base-2 softmax domain) ----
      // Batch all 4 K B-fragments so the ds loads pipeline ahead of WMMA.
      V16 bk00, bk01, bk10, bk11;
      {
        const unsigned short* kp0 = &ks[(kc + ln) * 72 + hs * 16];
        const unsigned short* kp1 = &ks[(kc + 16 + ln) * 72 + hs * 16];
        bk00.u[0] = *(const uint4*)(kp0);      bk00.u[1] = *(const uint4*)(kp0 + 8);
        bk10.u[0] = *(const uint4*)(kp0 + 32); bk10.u[1] = *(const uint4*)(kp0 + 40);
        bk01.u[0] = *(const uint4*)(kp1);      bk01.u[1] = *(const uint4*)(kp1 + 8);
        bk11.u[0] = *(const uint4*)(kp1 + 32); bk11.u[1] = *(const uint4*)(kp1 + 40);
      }
      v8f sc[2];
      sc[0] = wmma_bf16(qa[1], bk10, wmma_bf16(qa[0], bk00, v8f_zero()));
      sc[1] = wmma_bf16(qa[1], bk11, wmma_bf16(qa[0], bk01, v8f_zero()));

      const int mv0 = mk[kb + kc + ln];
      const int mv1 = mk[kb + kc + 16 + ln];
      #pragma unroll
      for (int j = 0; j < 8; ++j) {
        sc[0][j] = mv0 ? sc[0][j] * SCALE2 : MASKED2;
        sc[1][j] = mv1 ? sc[1][j] * SCALE2 : MASKED2;
      }

      // ---- online softmax; rows live on (VGPR j, 16-lane half-group) ----
      float p[2][8];
      #pragma unroll
      for (int j = 0; j < 8; ++j) {
        float tm = fmaxf(sc[0][j], sc[1][j]);
        #pragma unroll
        for (int mash = 1; mash < 16; mash <<= 1)
          tm = fmaxf(tm, __shfl_xor(tm, mash, 32));
        const float mn = fmaxf(mrow[j], tm);
        const float p0 = EXP2(sc[0][j] - mn);
        const float p1 = EXP2(sc[1][j] - mn);
        float rs = p0 + p1;
        #pragma unroll
        for (int mash = 1; mash < 16; mash <<= 1)
          rs += __shfl_xor(rs, mash, 32);
        const float corr = EXP2(mrow[j] - mn);
        lrow[j] = lrow[j] * corr + rs;
        mrow[j] = mn;
        #pragma unroll
        for (int dt = 0; dt < 4; ++dt) acc[dt][j] *= corr;
        p[0][j] = p0; p[1][j] = p1;
      }

      // ---- transpose P (C layout -> A layout) via private LDS region.
      // Same-wave LDS ops are in-order, so no barrier needed.
      #pragma unroll
      for (int ct = 0; ct < 2; ++ct)
        #pragma unroll
        for (int j = 0; j < 8; ++j)
          plds[wbase + (j + hs * 8) * 40 + ct * 16 + ln] = bfb(p[ct][j]);
      V16 pa;
      pa.u[0] = *(const uint4*)(&plds[wbase + ln * 40 + hs * 16]);
      pa.u[1] = *(const uint4*)(&plds[wbase + ln * 40 + hs * 16 + 8]);

      // ---- acc += P @ V (V pre-transposed: contiguous B fragments) ----
      V16 vb[4];
      #pragma unroll
      for (int dt = 0; dt < 4; ++dt) {
        const unsigned short* vp = &vs[(dt * 16 + ln) * 72 + kc + hs * 16];
        vb[dt].u[0] = *(const uint4*)(vp);
        vb[dt].u[1] = *(const uint4*)(vp + 8);
      }
      #pragma unroll
      for (int dt = 0; dt < 4; ++dt)
        acc[dt] = wmma_bf16(pa, vb[dt], acc[dt]);
    }
  }

  // ---- normalize and store bf16 attention output [token][E] ----
  #pragma unroll
  for (int j = 0; j < 8; ++j) {
    const float inv = 1.0f / lrow[j];
    const size_t t = (size_t)n * S + qBase + j + hs * 8;
    #pragma unroll
    for (int dt = 0; dt < 4; ++dt)
      Ab[t * E + h * D + dt * 16 + ln] = bfb(acc[dt][j] * inv);
  }
}

// ---------------------------------------------------------------------------
// Kernel 3: output projection, out = Ab @ Wo^T + bo. grid = (TOK/128, E/64).
// 64x512 Wo slice shared by all 8 waves: staged in LDS in 128-wide K panels.
// ---------------------------------------------------------------------------
__global__ __launch_bounds__(256) void oproj_kernel(
    const unsigned short* __restrict__ Ab, const unsigned short* __restrict__ Wob,
    const float* __restrict__ bo, float* __restrict__ out) {
  __shared__ __align__(16) unsigned short wlds[64 * 136];  // [eo][128+8 pad]

  const int lane = threadIdx.x & 31, wave = threadIdx.x >> 5;
  const int ln = lane & 15, hs = lane >> 4;
  const int rowBase = blockIdx.x * 128 + wave * 16;
  const int colBase = blockIdx.y * 64;

  v8f acc[4] = {v8f_zero(), v8f_zero(), v8f_zero(), v8f_zero()};
  const unsigned short* ap = Ab + (size_t)(rowBase + ln) * E;

  const int cr = threadIdx.x >> 2;          // 0..63 staging row (eo)
  const int cc = (threadIdx.x & 3) * 32;    // staging col part (32 halves)

  for (int kk0 = 0; kk0 < E; kk0 += 128) {
    __syncthreads();
    {
      const unsigned short* wsrc = Wob + (size_t)(colBase + cr) * E + kk0 + cc;
      #pragma unroll
      for (int i = 0; i < 4; ++i)
        STAGE16(&wlds[cr * 136 + cc + i * 8], wsrc + i * 8);
      STAGE_FENCE();
    }
    __syncthreads();

    #pragma unroll
    for (int kkl = 0; kkl < 4; ++kkl) {
      const int kk = kk0 + kkl * 32;
      V16 a;
      a.u[0] = *(const uint4*)(ap + kk + hs * 16);
      a.u[1] = *(const uint4*)(ap + kk + hs * 16 + 8);
      V16 b[4];
      #pragma unroll
      for (int c = 0; c < 4; ++c) {
        const unsigned short* wp = &wlds[(c * 16 + ln) * 136 + kkl * 32 + hs * 16];
        b[c].u[0] = *(const uint4*)(wp);
        b[c].u[1] = *(const uint4*)(wp + 8);
      }
      #pragma unroll
      for (int c = 0; c < 4; ++c)
        acc[c] = wmma_bf16(a, b[c], acc[c]);
    }
  }

  #pragma unroll
  for (int c = 0; c < 4; ++c) {
    const int e = colBase + c * 16 + ln;
    const float bb = bo[e];
    #pragma unroll
    for (int j = 0; j < 8; ++j)
      out[(size_t)(rowBase + j + hs * 8) * E + e] = acc[c][j] + bb;
  }
}

// ---------------------------------------------------------------------------
extern "C" void kernel_launch(void* const* d_in, const int* in_sizes, int n_in,
                              void* d_out, int out_size, void* d_ws, size_t ws_size,
                              hipStream_t stream) {
  (void)in_sizes; (void)n_in; (void)out_size; (void)ws_size;
  const float* q   = (const float*)d_in[0];
  const float* k   = (const float*)d_in[1];
  const float* v   = (const float*)d_in[2];
  const int*   msk = (const int*)d_in[3];
  const float* Wq  = (const float*)d_in[4];
  const float* bq  = (const float*)d_in[5];
  const float* Wk  = (const float*)d_in[6];
  const float* bk  = (const float*)d_in[7];
  const float* Wv  = (const float*)d_in[8];
  const float* bv  = (const float*)d_in[9];
  const float* Wo  = (const float*)d_in[10];
  const float* bo  = (const float*)d_in[11];
  float* out = (float*)d_out;

  // bf16 workspace: Qb,Kb (n,h,s,d) | Vt (n,h,d,s) | Ab (tok,e) | Wob (e,e)
  const size_t per = (size_t)NB * H * S * D;       // 4,194,304 halves each
  unsigned short* Qb  = (unsigned short*)d_ws;
  unsigned short* Kb  = Qb + per;
  unsigned short* Vt  = Kb + per;
  unsigned short* Ab  = Vt + per;                  // TOK*E == per
  unsigned short* Wob = Ab + per;                  // +262,144 halves

  cvt_wo_kernel<<<dim3((E * E) / 2048), 256, 0, stream>>>(Wo, Wob);
  qkv_proj_kernel<<<dim3(TOK / 128, H, 3), 256, 0, stream>>>(
      q, k, v, Wq, bq, Wk, bk, Wv, bv, Qb, Kb, Vt);
  attn_kernel<<<dim3(S / 128, NB * H), 256, 0, stream>>>(Qb, Kb, Vt, msk, Ab);
  oproj_kernel<<<dim3(TOK / 128, E / 64), 256, 0, stream>>>(Ab, Wob, bo, out);
}
